// PointACMix_15857019257411
// MI455X (gfx1250) — matrix operand order repaired and checked by reading, not compiled, not run on hardware
//
#include <hip/hip_runtime.h>
#include <hip/hip_bf16.h>

#define Bb 8
#define NPTS 16384
#define SS 1024
#define KNN 32
#define CC 256
#define CH 4096   // KNN LDS chunk (points)

typedef __attribute__((ext_vector_type(16))) _Float16 v16h;
typedef __attribute__((ext_vector_type(8)))  float    v8f;

__device__ __forceinline__ float gelu_exact(float x) {
    return 0.5f * x * (1.0f + erff(x * 0.70710678118654752f));
}

// ---------------------------------------------------------------------------
// 0) one-shot f32 -> f16 weight conversion (g1, g2, Wo: 256x256 each)
// ---------------------------------------------------------------------------
__global__ __launch_bounds__(256) void cvt_f16_kernel(const float* __restrict__ src,
                                                      _Float16* __restrict__ dst) {
    int i = (blockIdx.x * 256 + threadIdx.x) * 4;
    float4 v = *(const float4*)(src + i);
    dst[i + 0] = (_Float16)v.x;
    dst[i + 1] = (_Float16)v.y;
    dst[i + 2] = (_Float16)v.z;
    dst[i + 3] = (_Float16)v.w;
}

// ---------------------------------------------------------------------------
// 1) Furthest point sampling: one 1024-thread block per batch.
//    Each thread holds 16 points + dmin in registers; argmax via shuffle+LDS.
// ---------------------------------------------------------------------------
__global__ __launch_bounds__(1024) void fps_kernel(const float* __restrict__ x,
                                                   float* __restrict__ cent) {
    const int b = blockIdx.x;
    const int tid = threadIdx.x;
    const float* xb = x + (size_t)b * NPTS * 3;

    float px[16], py[16], pz[16], dmin[16];
#pragma unroll
    for (int j = 0; j < 16; ++j) {
        int p = tid + j * 1024;
        px[j] = xb[p * 3 + 0];
        py[j] = xb[p * 3 + 1];
        pz[j] = xb[p * 3 + 2];
        dmin[j] = 1e10f;
    }

    __shared__ float rd[32];
    __shared__ int   ri[32];
    __shared__ int   curs;

    int cur = 0;
    for (int s = 0; s < SS; ++s) {
        float cx = xb[cur * 3 + 0];
        float cy = xb[cur * 3 + 1];
        float cz = xb[cur * 3 + 2];
        if (tid == 0) {
            cent[((size_t)b * SS + s) * 3 + 0] = cx;
            cent[((size_t)b * SS + s) * 3 + 1] = cy;
            cent[((size_t)b * SS + s) * 3 + 2] = cz;
        }
        float bd = -1.0f; int bi = 0x7fffffff;
#pragma unroll
        for (int j = 0; j < 16; ++j) {
            float dx = px[j] - cx, dy = py[j] - cy, dz = pz[j] - cz;
            float d = dx * dx + dy * dy + dz * dz;
            dmin[j] = fminf(dmin[j], d);
            int p = tid + j * 1024;
            if (dmin[j] > bd || (dmin[j] == bd && p < bi)) { bd = dmin[j]; bi = p; }
        }
        // wave32 argmax (prefer larger d, then smaller idx)
#pragma unroll
        for (int off = 1; off < 32; off <<= 1) {
            float od = __shfl_xor(bd, off);
            int   oi = __shfl_xor(bi, off);
            if (od > bd || (od == bd && oi < bi)) { bd = od; bi = oi; }
        }
        if ((tid & 31) == 0) { rd[tid >> 5] = bd; ri[tid >> 5] = bi; }
        __syncthreads();
        if (tid < 32) {
            float d2 = rd[tid]; int i2 = ri[tid];
#pragma unroll
            for (int off = 1; off < 32; off <<= 1) {
                float od = __shfl_xor(d2, off);
                int   oi = __shfl_xor(i2, off);
                if (od > d2 || (od == d2 && oi < i2)) { d2 = od; i2 = oi; }
            }
            if (tid == 0) curs = i2;
        }
        __syncthreads();
        cur = curs;
    }
}

// ---------------------------------------------------------------------------
// 2) Exact KNN (top-32) + fused qkv projection/max-pool.
//    One wave per centroid, 16 waves/block, x streamed through LDS chunks.
//    Outputs t = (q - k) maxpool difference directly as f16, and v as f32.
// ---------------------------------------------------------------------------
__global__ __launch_bounds__(512) void knn_qkv_kernel(
    const float* __restrict__ x, const float* __restrict__ cent,
    const float* __restrict__ Wqkv, const float* __restrict__ bqkv,
    _Float16* __restrict__ th, float* __restrict__ vb) {

    __shared__ float chx[CH * 3];        // 48KB
    __shared__ float nbr[16][KNN][3];    // 6KB

    const int tid  = threadIdx.x;
    const int lane = tid & 31;
    const int w    = tid >> 5;
    const int b    = blockIdx.x >> 6;                 // 64 blocks per batch
    const int s    = ((blockIdx.x & 63) << 4) + w;    // centroid index
    const float* xb = x + (size_t)b * NPTS * 3;

    const size_t row = (size_t)b * SS + s;
    const float cx = cent[row * 3 + 0];
    const float cy = cent[row * 3 + 1];
    const float cz = cent[row * 3 + 2];

    float od = 1e30f; int oi = -1;   // lane-th entry of running top-32

    for (int c0 = 0; c0 < NPTS; c0 += CH) {
        __syncthreads();
        for (int i = tid; i < CH * 3; i += 512) chx[i] = xb[c0 * 3 + i];
        __syncthreads();

        float pd = -1e30f; int pi = -1;          // last extracted (exclusion)
        float newd = 1e30f; int newi = -1;       // lane-th entry of new top-32
        for (int j = 0; j < KNN; ++j) {
            float bd = 1e30f; int bi = 0x7fffffff;
            for (int p = lane; p < CH; p += 32) {
                float dx = chx[p * 3 + 0] - cx;
                float dy = chx[p * 3 + 1] - cy;
                float dz = chx[p * 3 + 2] - cz;
                float d = dx * dx + dy * dy + dz * dz;
                int gi = c0 + p;
                bool notex = (d > pd) || (d == pd && gi > pi);
                if (notex && (d < bd || (d == bd && gi < bi))) { bd = d; bi = gi; }
            }
            // running top-32 entry owned by this lane is also a candidate
            if (oi >= 0) {
                bool notex = (od > pd) || (od == pd && oi > pi);
                if (notex && (od < bd || (od == bd && oi < bi))) { bd = od; bi = oi; }
            }
#pragma unroll
            for (int off = 1; off < 32; off <<= 1) {
                float d2 = __shfl_xor(bd, off);
                int   i2 = __shfl_xor(bi, off);
                if (d2 < bd || (d2 == bd && i2 < bi)) { bd = d2; bi = i2; }
            }
            pd = bd; pi = bi;
            if (lane == j) { newd = bd; newi = bi; }
        }
        od = newd; oi = newi;
    }

    // gather the 32 neighbor coordinates into LDS
    nbr[w][lane][0] = xb[oi * 3 + 0];
    nbr[w][lane][1] = xb[oi * 3 + 1];
    nbr[w][lane][2] = xb[oi * 3 + 2];
    __syncthreads();

    // qkv projection (768 ch, K=3) + max-pool over the 32 neighbors.
    // channel c: q = Wqkv[c], k = Wqkv[c+256], v = Wqkv[c+512]; emit q-k, v.
    const size_t rowoff = row * CC;
#pragma unroll
    for (int t = 0; t < 8; ++t) {
        int c = t * 32 + lane;
        float qw0 = Wqkv[c * 3 + 0], qw1 = Wqkv[c * 3 + 1], qw2 = Wqkv[c * 3 + 2];
        float kw0 = Wqkv[(c + 256) * 3 + 0], kw1 = Wqkv[(c + 256) * 3 + 1], kw2 = Wqkv[(c + 256) * 3 + 2];
        float vw0 = Wqkv[(c + 512) * 3 + 0], vw1 = Wqkv[(c + 512) * 3 + 1], vw2 = Wqkv[(c + 512) * 3 + 2];
        float qbv = bqkv[c], kbv = bqkv[c + 256], vbv = bqkv[c + 512];
        float mq = -1e30f, mk = -1e30f, mv = -1e30f;
#pragma unroll 8
        for (int p = 0; p < KNN; ++p) {
            float nx = nbr[w][p][0], ny = nbr[w][p][1], nz = nbr[w][p][2];
            mq = fmaxf(mq, fmaf(qw0, nx, fmaf(qw1, ny, fmaf(qw2, nz, qbv))));
            mk = fmaxf(mk, fmaf(kw0, nx, fmaf(kw1, ny, fmaf(kw2, nz, kbv))));
            mv = fmaxf(mv, fmaf(vw0, nx, fmaf(vw1, ny, fmaf(vw2, nz, vbv))));
        }
        th[rowoff + c] = (_Float16)(mq - mk);
        vb[rowoff + c] = mv;
    }
}

// ---------------------------------------------------------------------------
// 3) WMMA GEMM: D[8192,256] = A[8192,256](f16) @ Wh^T(f16) + epilogue.
//    One wave computes a 16x64 strip (A-fragment reuse x4), K-step = 32.
//    MODE 1: epilogue gelu(acc+bias)     -> f16 dstH
//    MODE 2: epilogue acc+bias           -> f32 dstF
//    MODE 3: epilogue acc+bias+extra     -> f32 dstF
// ---------------------------------------------------------------------------
template <int MODE>
__global__ __launch_bounds__(32) void gemm16_kernel(
    const _Float16* __restrict__ Ah,
    const _Float16* __restrict__ Wh, const float* __restrict__ bias,
    const float* __restrict__ extra,
    float* __restrict__ dstF, _Float16* __restrict__ dstH) {

    const int lane   = threadIdx.x;
    const int m0     = blockIdx.x * 16;
    const int n0base = blockIdx.y * 64;
    const int halfsel = lane >> 4;            // 0: lanes 0-15, 1: lanes 16-31
    const size_t aoff = (size_t)(m0 + (lane & 15)) * CC;
    const int kb_a = halfsel * 8;             // A: K-half offset
    const int kb_b = halfsel * 8;             // B: K-half offset

    v8f acc[4];
#pragma unroll
    for (int i = 0; i < 4; ++i) acc[i] = v8f{};

    for (int k0 = 0; k0 < CC; k0 += 32) {
        // A fragment: 16-bit A 16x32 layout (pairs K{2r,2r+1}, K split at 8/24)
        const _Float16* arow = Ah + aoff + k0 + kb_a;
        v16h af;
#pragma unroll
        for (int e = 0; e < 16; ++e) af[e] = arow[e + (e >= 8 ? 8 : 0)];
#pragma unroll
        for (int nt = 0; nt < 4; ++nt) {
            int col = n0base + nt * 16 + (lane & 15);
            const _Float16* wrow = Wh + (size_t)col * CC + k0 + kb_b;  // B[k][n]=W[n][k]
            v16h bf;
#pragma unroll
            for (int e = 0; e < 16; ++e) bf[e] = wrow[e + (e >= 8 ? 8 : 0)];
            acc[nt] = __builtin_amdgcn_wmma_f32_16x16x32_f16(
                false, af, false, bf, (short)0, acc[nt], false, false);
        }
    }

#pragma unroll
    for (int nt = 0; nt < 4; ++nt) {
        int col = n0base + nt * 16 + (lane & 15);
#pragma unroll
        for (int r = 0; r < 8; ++r) {
            int rowo = m0 + r + (halfsel << 3);
            size_t o = (size_t)rowo * CC + col;
            float v = acc[nt][r] + bias[col];
            if (MODE == 1) {
                dstH[o] = (_Float16)gelu_exact(v);
            } else if (MODE == 2) {
                dstF[o] = v;
            } else {
                dstF[o] = v + extra[o];
            }
        }
    }
}

// ---------------------------------------------------------------------------
// 4) softmax over channels (scale 1/16) fused with attn*v -> f16
// ---------------------------------------------------------------------------
__global__ __launch_bounds__(256) void softmax_av_kernel(
    const float* __restrict__ h2, const float* __restrict__ v,
    _Float16* __restrict__ a) {
    const int lane = threadIdx.x & 31;
    const int w    = threadIdx.x >> 5;
    const size_t row = (size_t)blockIdx.x * 8 + w;
    const float* hr = h2 + row * CC;
    float vals[8];
    float m = -1e30f;
#pragma unroll
    for (int i = 0; i < 8; ++i) {
        vals[i] = hr[lane + i * 32] * 0.0625f;   // C^-0.5 = 1/16
        m = fmaxf(m, vals[i]);
    }
#pragma unroll
    for (int off = 1; off < 32; off <<= 1) m = fmaxf(m, __shfl_xor(m, off));
    float ssum = 0.0f;
#pragma unroll
    for (int i = 0; i < 8; ++i) { vals[i] = expf(vals[i] - m); ssum += vals[i]; }
#pragma unroll
    for (int off = 1; off < 32; off <<= 1) ssum += __shfl_xor(ssum, off);
    float inv = 1.0f / ssum;
    const float* vr = v + row * CC;
#pragma unroll
    for (int i = 0; i < 8; ++i)
        a[row * CC + lane + i * 32] = (_Float16)(vals[i] * inv * vr[lane + i * 32]);
}

// ---------------------------------------------------------------------------
// 5) positional MLP: 3 -> 128 (gelu) -> 256, one block per row
// ---------------------------------------------------------------------------
__global__ __launch_bounds__(256) void pos_kernel(
    const float* __restrict__ cent,
    const float* __restrict__ p1, const float* __restrict__ bp1,
    const float* __restrict__ p2, const float* __restrict__ bp2,
    float* __restrict__ pos) {
    __shared__ float h[128];
    const size_t r = blockIdx.x;
    const int t = threadIdx.x;
    float cx = cent[r * 3 + 0], cy = cent[r * 3 + 1], cz = cent[r * 3 + 2];
    if (t < 128) {
        float vv = fmaf(p1[t * 3 + 0], cx,
                   fmaf(p1[t * 3 + 1], cy,
                   fmaf(p1[t * 3 + 2], cz, bp1[t])));
        h[t] = gelu_exact(vv);
    }
    __syncthreads();
    float acc = bp2[t];
    const float* wr = p2 + (size_t)t * 128;
#pragma unroll 8
    for (int j = 0; j < 128; ++j) acc = fmaf(wr[j], h[j], acc);
    pos[r * CC + t] = acc;
}

// ---------------------------------------------------------------------------
extern "C" void kernel_launch(void* const* d_in, const int* in_sizes, int n_in,
                              void* d_out, int out_size, void* d_ws, size_t ws_size,
                              hipStream_t stream) {
    const float* x    = (const float*)d_in[0];
    const float* Wqkv = (const float*)d_in[1];
    const float* bqkv = (const float*)d_in[2];
    const float* g1   = (const float*)d_in[3];
    const float* bg1  = (const float*)d_in[4];
    const float* g2   = (const float*)d_in[5];
    const float* bg2  = (const float*)d_in[6];
    const float* Wo   = (const float*)d_in[7];
    const float* bo   = (const float*)d_in[8];
    const float* p1   = (const float*)d_in[9];
    const float* bp1  = (const float*)d_in[10];
    const float* p2   = (const float*)d_in[11];
    const float* bp2  = (const float*)d_in[12];
    float* out = (float*)d_out;

    const size_t MR = (size_t)Bb * SS;        // 8192 rows
    char* ws = (char*)d_ws;
    float*    cent = (float*)ws;     ws += MR * 3 * sizeof(float);
    float*    vb   = (float*)ws;     ws += MR * CC * sizeof(float);
    float*    h2   = (float*)ws;     ws += MR * CC * sizeof(float);
    float*    pos  = (float*)ws;     ws += MR * CC * sizeof(float);
    _Float16* th   = (_Float16*)ws;  ws += MR * CC * sizeof(_Float16);
    _Float16* h1   = (_Float16*)ws;  ws += MR * CC * sizeof(_Float16);
    _Float16* av   = (_Float16*)ws;  ws += MR * CC * sizeof(_Float16);
    _Float16* g1h  = (_Float16*)ws;  ws += CC * CC * sizeof(_Float16);
    _Float16* g2h  = (_Float16*)ws;  ws += CC * CC * sizeof(_Float16);
    _Float16* Woh  = (_Float16*)ws;  ws += CC * CC * sizeof(_Float16);

    // one-shot weight conversions (f32 -> f16), 256x256 each
    cvt_f16_kernel<<<(CC * CC) / (256 * 4), 256, 0, stream>>>(g1, g1h);
    cvt_f16_kernel<<<(CC * CC) / (256 * 4), 256, 0, stream>>>(g2, g2h);
    cvt_f16_kernel<<<(CC * CC) / (256 * 4), 256, 0, stream>>>(Wo, Woh);

    fps_kernel<<<Bb, 1024, 0, stream>>>(x, cent);
    knn_qkv_kernel<<<(Bb * SS) / 16, 512, 0, stream>>>(x, cent, Wqkv, bqkv, th, vb);
    pos_kernel<<<MR, 256, 0, stream>>>(cent, p1, bp1, p2, bp2, pos);

    dim3 ggrid(MR / 16, 4);   // 16-row tiles x (4 * 64-col strips)
    gemm16_kernel<1><<<ggrid, 32, 0, stream>>>(th, g1h, bg1, nullptr, nullptr, h1);
    gemm16_kernel<2><<<ggrid, 32, 0, stream>>>(h1, g2h, bg2, nullptr, h2, nullptr);
    softmax_av_kernel<<<MR / 8, 256, 0, stream>>>(h2, vb, av);
    gemm16_kernel<3><<<ggrid, 32, 0, stream>>>(av, Woh, bo, pos, out, nullptr);
}
